// BidirectionalCrossAttention_60610578481568
// MI455X (gfx1250) — matrix operand order repaired and checked by reading, not compiled, not run on hardware
//
#include <hip/hip_runtime.h>

// ---------------------------------------------------------------------------
// Types
// ---------------------------------------------------------------------------
typedef __attribute__((ext_vector_type(16))) __bf16 v16bf;
typedef __attribute__((ext_vector_type(8)))  float  v8f;
typedef __attribute__((ext_vector_type(4)))  unsigned int u32x4;
typedef __attribute__((ext_vector_type(4)))  float  f32x4;

union Frag {
    v16bf          v;
    unsigned short h[16];
    u32x4          q[2];
};
union PackBF8 {
    __bf16 b[8];
    u32x4  q;
};

__device__ __forceinline__ __bf16 f2bf(float f) {
    unsigned u = __builtin_bit_cast(unsigned, f);
    unsigned r = (u + 0x7FFFu + ((u >> 16) & 1u)) >> 16;
    unsigned short s = (unsigned short)r;
    return __builtin_bit_cast(__bf16, s);
}

#define DIMC   1024
#define HEADS  16
#define DHEAD  64
#define BATCH  4
#define NSEQ   1024
#define NKV2   2048

// ---------------------------------------------------------------------------
// Transposing f32 -> bf16 weight conversion: in (K x N) f32 -> out (N x K) bf16
// ---------------------------------------------------------------------------
__global__ __launch_bounds__(256) void tcvt_kernel(const float* __restrict__ in,
                                                   __bf16* __restrict__ out,
                                                   int K, int N) {
    __shared__ float t[32][33];
    int n0 = blockIdx.x * 32, k0 = blockIdx.y * 32;
    int tx = threadIdx.x, ty = threadIdx.y;  // 32 x 8
#pragma unroll
    for (int j = 0; j < 32; j += 8)
        t[ty + j][tx] = in[(size_t)(k0 + ty + j) * N + n0 + tx];
    __syncthreads();
#pragma unroll
    for (int j = 0; j < 32; j += 8)
        out[(size_t)(n0 + ty + j) * K + k0 + tx] = f2bf(t[tx][ty + j]);
}

// ---------------------------------------------------------------------------
// LayerNorm over rows of 1024. Optional outputs:
//   out1 (bf16) at ((batch*o1_bstride + o1_roff + r) * 1024 + c)  -- concat layout
//   out2 (bf16) plain row-major
//   outf (f32)  plain row-major
// ---------------------------------------------------------------------------
__global__ __launch_bounds__(256) void ln_kernel(const float* __restrict__ x,
                                                 const float* __restrict__ gam,
                                                 const float* __restrict__ bet,
                                                 __bf16* out1, int o1_bstride, int o1_roff,
                                                 __bf16* out2, float* outf,
                                                 int rows_per_batch) {
    __shared__ float s1[256];
    __shared__ float s2[256];
    int row = blockIdx.x;
    int tid = threadIdx.x;
    const float* xr = x + (size_t)row * DIMC;
    float v[4];
    float s = 0.f, sq = 0.f;
#pragma unroll
    for (int i = 0; i < 4; ++i) {
        v[i] = xr[tid + 256 * i];
        s += v[i];
        sq += v[i] * v[i];
    }
    s1[tid] = s; s2[tid] = sq;
    __syncthreads();
    for (int st = 128; st > 0; st >>= 1) {
        if (tid < st) { s1[tid] += s1[tid + st]; s2[tid] += s2[tid + st]; }
        __syncthreads();
    }
    float mu   = s1[0] * (1.f / DIMC);
    float var  = s2[0] * (1.f / DIMC) - mu * mu;
    float rstd = rsqrtf(var + 1e-5f);
    int batch = row / rows_per_batch;
    int r     = row % rows_per_batch;
    size_t o1base = (size_t)((size_t)batch * o1_bstride + o1_roff + r) * DIMC;
#pragma unroll
    for (int i = 0; i < 4; ++i) {
        int c = tid + 256 * i;
        float y = (v[i] - mu) * rstd * gam[c] + bet[c];
        if (out1) out1[o1base + c] = f2bf(y);
        if (out2) out2[(size_t)row * DIMC + c] = f2bf(y);
        if (outf) outf[(size_t)row * DIMC + c] = y;
    }
}

// ---------------------------------------------------------------------------
// bf16 WMMA GEMM computing C^T tiles:
//   C[m][n] = sum_k Aact[m][k] * Wt[n][k]  (+bias_n[n]) (+bias_m[m]) (+residual)
// Aact: (M x K) bf16 row-major, Wt: (N x K) bf16 row-major, C: (M x N)
// block tile: 128(N) x 128(M); 8 waves = 4 in N x 2 in M; wave tile 32(N) x 64(M)
// All fragment loads are contiguous ds_load_b128; stores vectorized.
// act: 0=none 1=relu 2=gelu(exact) 3=sigmoid
// ---------------------------------------------------------------------------
#define GLD 40
__global__ __launch_bounds__(256) void gemm_bf16_kernel(
    const __bf16* __restrict__ Aact, const __bf16* __restrict__ Wt,
    const float* __restrict__ bias_n, const float* __restrict__ bias_m,
    const float* __restrict__ residual,
    float* __restrict__ Cf, __bf16* __restrict__ Cbf,
    int M, int N, int K, int act) {
    __shared__ __attribute__((aligned(16))) __bf16 lW[128 * GLD];  // Wt rows (n)
    __shared__ __attribute__((aligned(16))) __bf16 lX[128 * GLD];  // Aact rows (m)

    int tid  = threadIdx.x;
    int wave = tid >> 5;
    int lane = tid & 31;
    int half = lane >> 4;
    int ln   = lane & 15;
    int wn   = (wave & 3) * 32;   // wave N offset in block tile
    int wm   = (wave >> 2) * 64;  // wave M offset in block tile
    int m0   = blockIdx.y * 128;
    int n0   = blockIdx.x * 128;

    v8f acc[2][4] = {};

    for (int kt = 0; kt < K; kt += 32) {
        // Stage Wt tile 128(n) x 32(k) and Aact tile 128(m) x 32(k): 512 chunks each
#pragma unroll
        for (int t2 = 0; t2 < 2; ++t2) {
            int c   = tid + 256 * t2;
            int row = c >> 2;
            int cc  = c & 3;
            *(u32x4*)(&lW[row * GLD + cc * 8]) =
                *(const u32x4*)(Wt + (size_t)(n0 + row) * K + kt + cc * 8);
            *(u32x4*)(&lX[row * GLD + cc * 8]) =
                *(const u32x4*)(Aact + (size_t)(m0 + row) * K + kt + cc * 8);
        }
        if (kt + 32 < K)
            __builtin_prefetch(Aact + (size_t)(m0 + (tid >> 1)) * K + kt + 32, 0, 3);
        __syncthreads();

        // A fragments (weights): lane row n = wn+i*16+ln, K runs [8h..+7],[8h+16..+23]
        Frag af[2];
#pragma unroll
        for (int i = 0; i < 2; ++i) {
            const __bf16* p = &lW[(wn + i * 16 + ln) * GLD + 8 * half];
            af[i].q[0] = *(const u32x4*)(p);
            af[i].q[1] = *(const u32x4*)(p + 16);
        }
        // B fragments (activation): lane col m = wm+t*16+ln, K runs [8h..+7],[8h+16..+23]
        Frag bfr[4];
#pragma unroll
        for (int t = 0; t < 4; ++t) {
            const __bf16* p = &lX[(wm + t * 16 + ln) * GLD + 8 * half];
            bfr[t].q[0] = *(const u32x4*)(p);
            bfr[t].q[1] = *(const u32x4*)(p + 16);
        }
#pragma unroll
        for (int i = 0; i < 2; ++i)
#pragma unroll
            for (int t = 0; t < 4; ++t)
                acc[i][t] = __builtin_amdgcn_wmma_f32_16x16x32_bf16(
                    false, af[i].v, false, bfr[t].v, (short)0, acc[i][t], false, false);
        __syncthreads();
    }

    // Epilogue: acc[i][t][r] -> n = n0+wn+i*16+8*half+r (contiguous in r), m fixed/lane
#pragma unroll
    for (int i = 0; i < 2; ++i) {
        int nb = n0 + wn + i * 16 + 8 * half;  // 8 contiguous n per lane
        f32x4 bn0 = {}, bn1 = {};
        if (bias_n) {
            bn0 = *(const f32x4*)(bias_n + nb);
            bn1 = *(const f32x4*)(bias_n + nb + 4);
        }
#pragma unroll
        for (int t = 0; t < 4; ++t) {
            int m = m0 + wm + t * 16 + ln;
            float bm = bias_m ? bias_m[m] : 0.f;
            float v[8];
#pragma unroll
            for (int r = 0; r < 8; ++r)
                v[r] = acc[i][t][r] + (r < 4 ? bn0[r] : bn1[r - 4]) + bm;
            if (residual) {
                f32x4 r0 = *(const f32x4*)(residual + (size_t)m * N + nb);
                f32x4 r1 = *(const f32x4*)(residual + (size_t)m * N + nb + 4);
#pragma unroll
                for (int r = 0; r < 8; ++r) v[r] += (r < 4 ? r0[r] : r1[r - 4]);
            }
#pragma unroll
            for (int r = 0; r < 8; ++r) {
                if (act == 1)      v[r] = fmaxf(v[r], 0.f);
                else if (act == 2) v[r] = 0.5f * v[r] * (1.f + erff(v[r] * 0.70710678118f));
                else if (act == 3) v[r] = 1.f / (1.f + __expf(-v[r]));
            }
            if (Cf) {
                f32x4 o0 = {v[0], v[1], v[2], v[3]};
                f32x4 o1 = {v[4], v[5], v[6], v[7]};
                *(f32x4*)(Cf + (size_t)m * N + nb)     = o0;
                *(f32x4*)(Cf + (size_t)m * N + nb + 4) = o1;
            }
            if (Cbf) {
                PackBF8 pk;
#pragma unroll
                for (int r = 0; r < 8; ++r) pk.b[r] = f2bf(v[r]);
                *(u32x4*)(Cbf + (size_t)m * N + nb) = pk.q;
            }
        }
    }
}

// ---------------------------------------------------------------------------
// Flash attention:
//   Q (B*NQ, 1024=h*64+d) bf16, K (B*NKV, 1024) bf16, Vt (1024, B*NKV) bf16
//   O (B*NQ, 1024) bf16.  grid (NQ/64, HEADS, B), block 128 (4 waves x 16 q rows)
// S = Q.K^T via WMMA (contiguous frags); O^T = V^T . P^T via WMMA
// (Vt rows contiguous in j; P^T frags contiguous from row-major ldsP).
// ---------------------------------------------------------------------------
#define PLD 40
__global__ __launch_bounds__(128) void attn_kernel(
    const __bf16* __restrict__ Q, const __bf16* __restrict__ K,
    const __bf16* __restrict__ Vt, __bf16* __restrict__ O,
    int NQ, int NKV, int ldv, float scale) {
    __shared__ __attribute__((aligned(16))) __bf16 lP[4 * 16 * PLD];
    __shared__ float lAux[4][16];

    int tid  = threadIdx.x;
    int wave = tid >> 5;
    int lane = tid & 31;
    int half = lane >> 4;
    int ln   = lane & 15;
    int hd   = blockIdx.y;
    int b    = blockIdx.z;
    int qbase = blockIdx.x * 64 + wave * 16;
    size_t qrow = (size_t)b * NQ + qbase;

    // Q fragments (A-layout): row = ln, K runs [8h..+7],[8h+16..+23] within 32-d window
    Frag qf[2];
#pragma unroll
    for (int dh = 0; dh < 2; ++dh) {
        const __bf16* p = Q + (qrow + ln) * DIMC + hd * DHEAD + dh * 32 + 8 * half;
        qf[dh].q[0] = *(const u32x4*)(p);
        qf[dh].q[1] = *(const u32x4*)(p + 16);
    }

    float mrow[8], lrow[8];
#pragma unroll
    for (int r = 0; r < 8; ++r) { mrow[r] = -1e30f; lrow[r] = 0.f; }
    v8f oacc[4] = {};  // O^T: rows d = t*16 + 8*half + r, col m = ln
    __bf16* myP = lP + wave * 16 * PLD;

    for (int jt = 0; jt < NKV; jt += 32) {
        // S tiles: S[s] rows = q (r+8h), cols = j (ln); contract over d
        v8f S[2] = {};
#pragma unroll
        for (int s = 0; s < 2; ++s) {
#pragma unroll
            for (int dh = 0; dh < 2; ++dh) {
                Frag kf;  // B-layout: lane col j = ln, K(d) runs [8h..+7],[8h+16..+23]
                const __bf16* p = K + ((size_t)b * NKV + jt + s * 16 + ln) * DIMC +
                                  hd * DHEAD + dh * 32 + 8 * half;
                kf.q[0] = *(const u32x4*)(p);
                kf.q[1] = *(const u32x4*)(p + 16);
                S[s] = __builtin_amdgcn_wmma_f32_16x16x32_bf16(
                    false, qf[dh].v, false, kf.v, (short)0, S[s], false, false);
            }
        }

        // Online softmax over rows r+8h (each row lives across 16 lanes of a half)
        float p0[8], p1[8], alpha[8];
#pragma unroll
        for (int r = 0; r < 8; ++r) {
            float sm = fmaxf(S[0][r], S[1][r]) * scale;
#pragma unroll
            for (int msk = 1; msk < 16; msk <<= 1) sm = fmaxf(sm, __shfl_xor(sm, msk, 32));
            float mnew = fmaxf(mrow[r], sm);
            alpha[r] = __expf(mrow[r] - mnew);
            p0[r] = __expf(S[0][r] * scale - mnew);
            p1[r] = __expf(S[1][r] * scale - mnew);
            float rs = p0[r] + p1[r];
#pragma unroll
            for (int msk = 1; msk < 16; msk <<= 1) rs += __shfl_xor(rs, msk, 32);
            lrow[r] = lrow[r] * alpha[r] + rs;
            mrow[r] = mnew;
        }

        // Store P row-major 16x32 (padded), publish per-row alpha for O^T lanes
#pragma unroll
        for (int r = 0; r < 8; ++r) {
            myP[(r + 8 * half) * PLD + ln]      = f2bf(p0[r]);
            myP[(r + 8 * half) * PLD + 16 + ln] = f2bf(p1[r]);
        }
        if (ln == 0) {
#pragma unroll
            for (int r = 0; r < 8; ++r) lAux[wave][8 * half + r] = alpha[r];
        }
        __syncthreads();

        float alphaLane = lAux[wave][ln];
#pragma unroll
        for (int t = 0; t < 4; ++t)
#pragma unroll
            for (int r = 0; r < 8; ++r) oacc[t][r] *= alphaLane;

        // P^T B-fragment: lane col m = ln, K(j) runs [8h..+7],[8h+16..+23]
        Frag pf;
        {
            const __bf16* pp = myP + ln * PLD + 8 * half;
            pf.q[0] = *(const u32x4*)(pp);
            pf.q[1] = *(const u32x4*)(pp + 16);
        }
        // V^T A-fragments: lane row d = t*16+ln, K(j) runs [8h..+7],[8h+16..+23]
#pragma unroll
        for (int t = 0; t < 4; ++t) {
            Frag vf;
            const __bf16* p = Vt + (size_t)(hd * DHEAD + t * 16 + ln) * ldv +
                              (size_t)b * NKV + jt + 8 * half;
            vf.q[0] = *(const u32x4*)(p);
            vf.q[1] = *(const u32x4*)(p + 16);
            oacc[t] = __builtin_amdgcn_wmma_f32_16x16x32_bf16(
                false, vf.v, false, pf.v, (short)0, oacc[t], false, false);
        }
        __syncthreads();
    }

    // Publish per-row l, then vectorized store of O (8 contiguous d per lane)
    if (ln == 0) {
#pragma unroll
        for (int r = 0; r < 8; ++r) lAux[wave][8 * half + r] = lrow[r];
    }
    __syncthreads();
    float inv = 1.f / lAux[wave][ln];
#pragma unroll
    for (int t = 0; t < 4; ++t) {
        PackBF8 pk;
#pragma unroll
        for (int r = 0; r < 8; ++r) pk.b[r] = f2bf(oacc[t][r] * inv);
        *(u32x4*)(O + (qrow + ln) * DIMC + hd * DHEAD + t * 16 + 8 * half) = pk.q;
    }
}

// ---------------------------------------------------------------------------
// mean over sequence axis: x (4,1024,1024) -> out (4,1024)
// ---------------------------------------------------------------------------
__global__ __launch_bounds__(256) void mean_kernel(const float* __restrict__ x,
                                                   float* __restrict__ out) {
    int idx = blockIdx.x * 256 + threadIdx.x;  // 0 .. 4095
    int b = idx >> 10, d = idx & 1023;
    float s = 0.f;
    for (int n = 0; n < NSEQ; ++n) s += x[((size_t)b * NSEQ + n) * DIMC + d];
    out[idx] = s * (1.f / NSEQ);
}

// fusion_in = concat([fwd_out, broadcast(rev_global)], -1) as bf16, (4096, 2048)
__global__ __launch_bounds__(256) void fusin_kernel(const float* __restrict__ fwd,
                                                    const float* __restrict__ revg,
                                                    __bf16* __restrict__ out) {
    size_t idx = (size_t)blockIdx.x * 256 + threadIdx.x;
    int row = (int)(idx >> 11);
    int col = (int)(idx & 2047);
    int b = row >> 10;
    float v = (col < DIMC) ? fwd[(size_t)row * DIMC + col]
                           : revg[b * DIMC + (col - DIMC)];
    out[idx] = f2bf(v);
}

// gated = gate*fwd + (1-gate)*rev_exp ; fcat = concat([gated, fwd]) bf16 (4096, 2048)
__global__ __launch_bounds__(256) void gated_kernel(const float* __restrict__ gate,
                                                    const float* __restrict__ fwd,
                                                    const float* __restrict__ revg,
                                                    __bf16* __restrict__ fcat) {
    size_t idx = (size_t)blockIdx.x * 256 + threadIdx.x;  // 0 .. 4096*1024-1
    int row = (int)(idx >> 10);
    int col = (int)(idx & 1023);
    int b = row >> 10;
    float gt = gate[idx];
    float fv = fwd[idx];
    float rv = revg[b * DIMC + col];
    fcat[(size_t)row * 2048 + col]        = f2bf(gt * fv + (1.f - gt) * rv);
    fcat[(size_t)row * 2048 + DIMC + col] = f2bf(fv);
}

// ---------------------------------------------------------------------------
// Host-side orchestration
// ---------------------------------------------------------------------------
extern "C" void kernel_launch(void* const* d_in, const int* in_sizes, int n_in,
                              void* d_out, int out_size, void* d_ws, size_t ws_size,
                              hipStream_t stream) {
    (void)in_sizes; (void)n_in; (void)out_size; (void)ws_size;
    const float* query = (const float*)d_in[0];
    const float* kv    = (const float*)d_in[1];
    // fwd params at 2..13, rev at 14..25, then globals 26..39
    auto F = [&](int i) { return (const float*)d_in[i]; };

    char* ws = (char*)d_ws;
    size_t off = 0;
    auto alloc = [&](size_t bytes) -> char* {
        off = (off + 255) & ~(size_t)255;
        char* p = ws + off;
        off += bytes;
        return p;
    };
    const size_t M1 = (size_t)1024 * 1024;  // 1M elems

    // Transposed bf16 weights (N x K layout)
    __bf16* wqTF = (__bf16*)alloc(M1 * 2);
    __bf16* wkTF = (__bf16*)alloc(M1 * 2);
    __bf16* wvTF = (__bf16*)alloc(M1 * 2);
    __bf16* woTF = (__bf16*)alloc(M1 * 2);
    __bf16* wqTR = (__bf16*)alloc(M1 * 2);
    __bf16* wkTR = (__bf16*)alloc(M1 * 2);
    __bf16* wvTR = (__bf16*)alloc(M1 * 2);
    __bf16* woTR = (__bf16*)alloc(M1 * 2);
    __bf16* gw1T = (__bf16*)alloc(4 * M1 * 2);
    __bf16* gw2T = (__bf16*)alloc(2 * M1 * 2);
    __bf16* fswT = (__bf16*)alloc(2 * M1 * 2);
    __bf16* fw1T = (__bf16*)alloc(2 * M1 * 2);
    __bf16* fw2T = (__bf16*)alloc(2 * M1 * 2);
    // activations
    __bf16* kvcat = (__bf16*)alloc(8 * M1 * 2);  // (B*2048, 1024)  [reused: fusion_in]
    __bf16* qn    = (__bf16*)alloc(4 * M1 * 2);  // (B*1024, 1024)  [reused: ffn h_in]
    __bf16* qb    = (__bf16*)alloc(4 * M1 * 2);
    __bf16* kb    = (__bf16*)alloc(8 * M1 * 2);  // [reused: g_bf, ffn mid]
    __bf16* vtb   = (__bf16*)alloc(8 * M1 * 2);  // V^T (1024, 8192) [reused: fused_cat]
    __bf16* attnb = (__bf16*)alloc(4 * M1 * 2);
    float* fwd_out = (float*)alloc(4 * M1 * 4);
    float* rev_out = (float*)alloc(4 * M1 * 4);
    float* revg    = (float*)alloc(4096 * 4);
    float* gate_f  = (float*)alloc(4 * M1 * 4);  // [reused: fused_pre]
    float* fused_f = (float*)alloc(4 * M1 * 4);

    // --- weight conversion (transpose + bf16) ---
    auto tcvt = [&](const float* src, __bf16* dst, int K, int N) {
        tcvt_kernel<<<dim3(N / 32, K / 32), dim3(32, 8), 0, stream>>>(src, dst, K, N);
    };
    tcvt(F(6),  wqTF, 1024, 1024);  tcvt(F(8),  wkTF, 1024, 1024);
    tcvt(F(10), wvTF, 1024, 1024);  tcvt(F(12), woTF, 1024, 1024);
    tcvt(F(18), wqTR, 1024, 1024);  tcvt(F(20), wkTR, 1024, 1024);
    tcvt(F(22), wvTR, 1024, 1024);  tcvt(F(24), woTR, 1024, 1024);
    tcvt(F(26), gw1T, 2048, 2048);
    tcvt(F(28), gw2T, 2048, 1024);
    tcvt(F(30), fswT, 2048, 1024);
    tcvt(F(36), fw1T, 1024, 2048);
    tcvt(F(38), fw2T, 2048, 1024);

    dim3 blk(256);
    auto gemm = [&](const __bf16* A, const __bf16* Wt, const float* bias_n,
                    const float* bias_m, const float* res, float* Cf, __bf16* Cbf,
                    int M, int N, int K, int act) {
        dim3 grid(N / 128, M / 128);
        gemm_bf16_kernel<<<grid, blk, 0, stream>>>(A, Wt, bias_n, bias_m, res, Cf, Cbf,
                                                   M, N, K, act);
    };

    // === fwd cross attention: query attends to [LN(kv); LN(query)] ===
    ln_kernel<<<4096, blk, 0, stream>>>(kv,    F(4), F(5), kvcat, 2048, 0,    nullptr, nullptr, 1024);
    ln_kernel<<<4096, blk, 0, stream>>>(query, F(2), F(3), kvcat, 2048, 1024, qn,      nullptr, 1024);
    gemm(qn,    wqTF, F(7),  nullptr, nullptr, nullptr, qb, 4096, 1024, 1024, 0);
    gemm(kvcat, wkTF, F(9),  nullptr, nullptr, nullptr, kb, 8192, 1024, 1024, 0);
    // V^T: (1024 x 8192) = wvT (as activation, M=1024) x kvcat (as Wt, N=8192)
    gemm(wvTF, kvcat, nullptr, F(11), nullptr, nullptr, vtb, 1024, 8192, 1024, 0);
    attn_kernel<<<dim3(16, 16, 4), dim3(128), 0, stream>>>(qb, kb, vtb, attnb,
                                                           NSEQ, NKV2, 8192, 0.125f);
    gemm(attnb, woTF, F(13), nullptr, query, fwd_out, nullptr, 4096, 1024, 1024, 0);

    // === rev cross attention: kv attends to [LN(query); LN(kv)] ===
    ln_kernel<<<4096, blk, 0, stream>>>(query, F(16), F(17), kvcat, 2048, 0,    nullptr, nullptr, 1024);
    ln_kernel<<<4096, blk, 0, stream>>>(kv,    F(14), F(15), kvcat, 2048, 1024, qn,      nullptr, 1024);
    gemm(qn,    wqTR, F(19), nullptr, nullptr, nullptr, qb, 4096, 1024, 1024, 0);
    gemm(kvcat, wkTR, F(21), nullptr, nullptr, nullptr, kb, 8192, 1024, 1024, 0);
    gemm(wvTR, kvcat, nullptr, F(23), nullptr, nullptr, vtb, 1024, 8192, 1024, 0);
    attn_kernel<<<dim3(16, 16, 4), dim3(128), 0, stream>>>(qb, kb, vtb, attnb,
                                                           NSEQ, NKV2, 8192, 0.125f);
    gemm(attnb, woTR, F(25), nullptr, kv, rev_out, nullptr, 4096, 1024, 1024, 0);

    // === fusion / gating ===
    mean_kernel<<<16, blk, 0, stream>>>(rev_out, revg);
    __bf16* fus_in = kvcat;  // reuse (B*1024, 2048)
    fusin_kernel<<<32768, blk, 0, stream>>>(fwd_out, revg, fus_in);
    __bf16* g_bf = kb;  // reuse
    gemm(fus_in, gw1T, F(27), nullptr, nullptr, nullptr, g_bf, 4096, 2048, 2048, 1);  // relu
    gemm(g_bf,   gw2T, F(29), nullptr, nullptr, gate_f, nullptr, 4096, 1024, 2048, 3);  // sigmoid
    __bf16* fcat = vtb;  // reuse (B*1024, 2048)
    gated_kernel<<<16384, blk, 0, stream>>>(gate_f, fwd_out, revg, fcat);
    float* fpre = gate_f;  // reuse
    gemm(fcat, fswT, F(31), nullptr, query, fpre, nullptr, 4096, 1024, 2048, 0);
    ln_kernel<<<4096, blk, 0, stream>>>(fpre, F(32), F(33), nullptr, 0, 0, nullptr, fused_f, 1024);

    // === FFN ===
    __bf16* hin = qn;  // reuse
    ln_kernel<<<4096, blk, 0, stream>>>(fused_f, F(34), F(35), nullptr, 0, 0, hin, nullptr, 1024);
    __bf16* mid = kb;  // reuse
    gemm(hin, fw1T, F(37), nullptr, nullptr, nullptr, mid, 4096, 2048, 1024, 2);  // gelu
    gemm(mid, fw2T, F(39), nullptr, fused_f, (float*)d_out, nullptr, 4096, 1024, 2048, 0);
}
